// Glm4MoE_75247827026061
// MI455X (gfx1250) — matrix operand order repaired and checked by reading, not compiled, not run on hardware
//
#include <hip/hip_runtime.h>
#include <hip/hip_bf16.h>

// ---------------- problem constants ----------------
constexpr int kT = 1024;   // tokens
constexpr int kH = 2048;   // hidden
constexpr int kI = 1024;   // intermediate
constexpr int kE = 16;     // experts
constexpr int kTopK = 4;
constexpr float kRoutedScale = 2.5f;

constexpr int kLdsU = 18;  // LDS row stride in uints (36 bf16, 72B: 8B-aligned, padded)

// ---------------- WMMA vector types ----------------
typedef __attribute__((ext_vector_type(16))) __bf16 v16bf;
typedef __attribute__((ext_vector_type(8)))  float  v8f;
typedef __attribute__((ext_vector_type(4)))  float  v4f;

__device__ inline unsigned pack_bf16(float a, float b)
{
    union { __bf16 h[2]; unsigned u; } r;
    r.h[0] = (__bf16)a; r.h[1] = (__bf16)b;
    return r.u;   // lowers to v_cvt_pk_bf16_f32
}

// =====================================================================
// Kernel 1: router.  One wave (32 lanes) per token.  E=16 dot products
// over H=2048, shuffle-reduced; grouped top-k done serially on lane 0.
// =====================================================================
__global__ void __launch_bounds__(128)
router_kernel(const float* __restrict__ x, const float* __restrict__ gw,
              const float* __restrict__ eb,
              int* __restrict__ topk_ids, float* __restrict__ topk_w,
              int* __restrict__ counts)
{
    if (blockIdx.x == 0 && threadIdx.x < kE) counts[threadIdx.x] = 0;

    const int wave = threadIdx.x >> 5;
    const int lane = threadIdx.x & 31;
    const int t = blockIdx.x * 4 + wave;
    if (t >= kT) return;

    float acc[kE];
#pragma unroll
    for (int e = 0; e < kE; ++e) acc[e] = 0.f;

    for (int h = lane; h < kH; h += 32) {
        const float xv = x[(size_t)t * kH + h];
#pragma unroll
        for (int e = 0; e < kE; ++e) acc[e] += xv * gw[(size_t)e * kH + h];
    }
#pragma unroll
    for (int e = 0; e < kE; ++e)
        for (int off = 16; off > 0; off >>= 1)
            acc[e] += __shfl_xor(acc[e], off, 32);

    if (lane != 0) return;

    float sc[kE], sb[kE];
#pragma unroll
    for (int e = 0; e < kE; ++e) {
        const float s = 1.f / (1.f + __expf(-acc[e]));
        sc[e] = s;
        sb[e] = s + eb[e];
    }
    // group score = sum of top-2 biased scores within each group of 4
    float gs[4];
#pragma unroll
    for (int g = 0; g < 4; ++g) {
        float m1 = -1e30f, m2 = -1e30f;
#pragma unroll
        for (int j = 0; j < 4; ++j) {
            const float v = sb[g * 4 + j];
            if (v > m1) { m2 = m1; m1 = v; } else if (v > m2) { m2 = v; }
        }
        gs[g] = m1 + m2;
    }
    int g1 = 0;
    for (int g = 1; g < 4; ++g) if (gs[g] > gs[g1]) g1 = g;
    int g2 = -1;
    for (int g = 0; g < 4; ++g) { if (g == g1) continue; if (g2 < 0 || gs[g] > gs[g2]) g2 = g; }

    float tmp[kE];
#pragma unroll
    for (int e = 0; e < kE; ++e) {
        const int g = e >> 2;
        tmp[e] = (g == g1 || g == g2) ? sb[e] : 0.0f;   // matches jnp.where(..., 0.0)
    }
    int ids[kTopK]; float ws[kTopK]; float wsum = 0.f;
#pragma unroll
    for (int k = 0; k < kTopK; ++k) {
        int best = 0;
        for (int e = 1; e < kE; ++e) if (tmp[e] > tmp[best]) best = e;
        ids[k] = best; ws[k] = sc[best]; wsum += sc[best];
        tmp[best] = -1e30f;
    }
    const float inv = 1.f / wsum;
#pragma unroll
    for (int k = 0; k < kTopK; ++k) {
        topk_ids[t * kTopK + k] = ids[k];
        topk_w[t * kTopK + k]   = ws[k] * inv;
    }
}

// =====================================================================
// Kernel 2: compact per-expert token lists (order irrelevant to output).
// =====================================================================
__global__ void __launch_bounds__(256)
compact_kernel(const int* __restrict__ topk_ids, int* __restrict__ counts,
               int* __restrict__ list_t, int* __restrict__ list_k)
{
    const int t = blockIdx.x * blockDim.x + threadIdx.x;
    if (t >= kT) return;
#pragma unroll
    for (int k = 0; k < kTopK; ++k) {
        const int e = topk_ids[t * kTopK + k];
        const int pos = atomicAdd(&counts[e], 1);
        list_t[e * kT + pos] = t;
        list_k[e * kT + pos] = k;
    }
}

// ---------------------------------------------------------------------
// A-fragment gather from an LDS bf16 tile (rows of kLdsU uints = 36 bf16)
// A k order used here (contiguous, same as B): lane m=l&15; lanes 0-15 hold K 0..15,
// lanes 16-31 hold K 16..31; VGPR v holds K pair {2v,2v+1}+base.
// ---------------------------------------------------------------------
__device__ inline v16bf load_a_frag(const unsigned (*As)[kLdsU], int m, int kh8)
{
    const __bf16* rowp = (const __bf16*)As[m];
    v16bf a;
#pragma unroll
    for (int j = 0; j < 16; ++j) {
        const int vv = j >> 1, p = j & 1;
        const int kk = 2 * kh8 + 2 * vv + p;
        a[j] = rowp[kk];
    }
    return a;
}

// B-fragment: lane = column n (l&15); lanes 0-15 K=0..15, lanes 16-31
// K=16..31, contiguous per lane -> one 64B contiguous fp32 run.
__device__ inline v16bf load_b_frag(const float* __restrict__ p)
{
    const v4f* p4 = (const v4f*)p;
    v16bf b;
#pragma unroll
    for (int jj = 0; jj < 4; ++jj) {
        const v4f q = p4[jj];
        b[4 * jj + 0] = (__bf16)q[0];
        b[4 * jj + 1] = (__bf16)q[1];
        b[4 * jj + 2] = (__bf16)q[2];
        b[4 * jj + 3] = (__bf16)q[3];
    }
    return b;
}

// =====================================================================
// Kernel 3: gate_up GEMM + fused SiLU*mul -> bf16 activations.
// Block: 256 thr = 8 waves; M-tile = 32 gathered rows (two 16x16 WMMA
// row tiles sharing each B fragment -> 4 WMMA per K step per wave).
// Wave w owns cols [col, col+16) of I: gate row col+n, up row I+col+n.
// =====================================================================
__global__ void __launch_bounds__(256)
gu_kernel(const float* __restrict__ x, const float* __restrict__ wgu_all,
          const int* __restrict__ list_t, const int* __restrict__ list_k,
          const int* __restrict__ counts, __bf16* __restrict__ act,
          int shared_mode)
{
    __shared__ unsigned As[32][kLdsU];

    const int e    = blockIdx.z;
    const int cnt  = shared_mode ? kT : counts[e];
    const int row0 = blockIdx.y * 32;
    if (row0 >= cnt) return;

    const float* wgu = wgu_all + (shared_mode ? (size_t)0 : (size_t)e * (2 * kI) * kH);
    const int wave = threadIdx.x >> 5;
    const int lane = threadIdx.x & 31;
    const int col  = blockIdx.x * 128 + wave * 16;     // within [0, I)
    const int n    = lane & 15;
    const int kh8  = (lane >> 4) * 8;
    const int kh16 = (lane >> 4) * 16;

    // ---- staging assignment (fixed per thread): row sr, 4 cols at sc ----
    const int sr = threadIdx.x >> 3;          // 0..31
    const int sc = (threadIdx.x & 7) * 4;     // 0,4,...,28
    const int srClamped = min(row0 + sr, cnt - 1);
    const int stok = shared_mode ? srClamped : list_t[e * kT + srClamped];
    const float* srcRow = x + (size_t)stok * kH;   // hoisted out of K loop

    const float* bg = wgu + (size_t)(col + n) * kH + kh16;        // gate row
    const float* bu = wgu + (size_t)(kI + col + n) * kH + kh16;   // up row

    v8f cg0 = {}, cg1 = {}, cu0 = {}, cu1 = {};

    for (int k0 = 0; k0 < kH; k0 += 32) {
        // stage A tile fp32 -> packed bf16, one b128 load + one b64 LDS store
        const v4f q = *(const v4f*)(srcRow + k0 + sc);
        uint2 pk;
        pk.x = pack_bf16(q[0], q[1]);
        pk.y = pack_bf16(q[2], q[3]);
        *(uint2*)&As[sr][sc >> 1] = pk;
        __syncthreads();

        const v16bf a0  = load_a_frag(As,      lane & 15, kh8);
        const v16bf a1  = load_a_frag(As + 16, lane & 15, kh8);
        const v16bf bgf = load_b_frag(bg + k0);
        const v16bf buf = load_b_frag(bu + k0);

        cg0 = __builtin_amdgcn_wmma_f32_16x16x32_bf16(false, a0, false, bgf, (short)0, cg0, false, false);
        cg1 = __builtin_amdgcn_wmma_f32_16x16x32_bf16(false, a1, false, bgf, (short)0, cg1, false, false);
        cu0 = __builtin_amdgcn_wmma_f32_16x16x32_bf16(false, a0, false, buf, (short)0, cu0, false, false);
        cu1 = __builtin_amdgcn_wmma_f32_16x16x32_bf16(false, a1, false, buf, (short)0, cu1, false, false);
        __syncthreads();
    }

    // D layout: VGPR r -> row r (+8 for lanes 16-31), col = lane&15
#pragma unroll
    for (int tile = 0; tile < 2; ++tile) {
        const v8f& g8 = tile ? cg1 : cg0;
        const v8f& u8 = tile ? cu1 : cu0;
#pragma unroll
        for (int r = 0; r < 8; ++r) {
            const int m = tile * 16 + r + (lane >> 4) * 8;
            const int ridx = row0 + m;
            if (ridx < cnt) {
                const float g = g8[r], u = u8[r];
                const float s = g / (1.f + __expf(-g));         // silu
                const int arow = shared_mode ? ridx
                    : (list_t[e * kT + ridx] * kTopK + list_k[e * kT + ridx]);
                act[(size_t)arow * kI + col + n] = (__bf16)(s * u);
            }
        }
    }
}

// =====================================================================
// Kernel 4: down GEMM (K = I), fp32 partial output per (t,k) row.
// Same skeleton: M=32, B fragment reused by two WMMAs.
// =====================================================================
__global__ void __launch_bounds__(256)
down_kernel(const __bf16* __restrict__ act, const float* __restrict__ wd_all,
            const int* __restrict__ list_t, const int* __restrict__ list_k,
            const int* __restrict__ counts, float* __restrict__ outp,
            int shared_mode)
{
    __shared__ unsigned As[32][kLdsU];

    const int e    = blockIdx.z;
    const int cnt  = shared_mode ? kT : counts[e];
    const int row0 = blockIdx.y * 32;
    if (row0 >= cnt) return;

    const float* wd = wd_all + (shared_mode ? (size_t)0 : (size_t)e * kH * kI);
    const int wave = threadIdx.x >> 5;
    const int lane = threadIdx.x & 31;
    const int col  = blockIdx.x * 128 + wave * 16;     // within [0, H)
    const int n    = lane & 15;
    const int kh8  = (lane >> 4) * 8;
    const int kh16 = (lane >> 4) * 16;

    // staging: row sr, 4 bf16 at col sc (already bf16 -> pure uint2 copy)
    const int sr = threadIdx.x >> 3;
    const int sc = (threadIdx.x & 7) * 4;
    const int srClamped = min(row0 + sr, cnt - 1);
    const int sarow = shared_mode ? srClamped
        : (list_t[e * kT + srClamped] * kTopK + list_k[e * kT + srClamped]);
    const __bf16* srcRow = act + (size_t)sarow * kI;

    const float* bp = wd + (size_t)(col + n) * kI + kh16;

    v8f c0 = {}, c1 = {};

    for (int k0 = 0; k0 < kI; k0 += 32) {
        *(uint2*)&As[sr][sc >> 1] = *(const uint2*)(srcRow + k0 + sc);
        __syncthreads();

        const v16bf a0 = load_a_frag(As,      lane & 15, kh8);
        const v16bf a1 = load_a_frag(As + 16, lane & 15, kh8);
        const v16bf b  = load_b_frag(bp + k0);

        c0 = __builtin_amdgcn_wmma_f32_16x16x32_bf16(false, a0, false, b, (short)0, c0, false, false);
        c1 = __builtin_amdgcn_wmma_f32_16x16x32_bf16(false, a1, false, b, (short)0, c1, false, false);
        __syncthreads();
    }

#pragma unroll
    for (int tile = 0; tile < 2; ++tile) {
        const v8f& cc = tile ? c1 : c0;
#pragma unroll
        for (int r = 0; r < 8; ++r) {
            const int m = tile * 16 + r + (lane >> 4) * 8;
            const int ridx = row0 + m;
            if (ridx < cnt) {
                const int arow = shared_mode ? ridx
                    : (list_t[e * kT + ridx] * kTopK + list_k[e * kT + ridx]);
                outp[(size_t)arow * kH + col + n] = cc[r];
            }
        }
    }
}

// =====================================================================
// Kernel 5: deterministic combine: out = shared + 2.5 * sum_k w_k*partial
// =====================================================================
__global__ void __launch_bounds__(256)
combine_kernel(const float* __restrict__ shared_out,
               const float* __restrict__ partial,
               const float* __restrict__ topk_w,
               float* __restrict__ out)
{
    const int idx = blockIdx.x * blockDim.x + threadIdx.x;   // over T*H
    const int t = idx / kH;
    const int h = idx - t * kH;
    float acc = 0.f;
#pragma unroll
    for (int k = 0; k < kTopK; ++k)
        acc += topk_w[t * kTopK + k] * partial[(size_t)(t * kTopK + k) * kH + h];
    out[idx] = acc * kRoutedScale + shared_out[idx];
}

// =====================================================================
// Host launcher
// =====================================================================
extern "C" void kernel_launch(void* const* d_in, const int* in_sizes, int n_in,
                              void* d_out, int out_size, void* d_ws, size_t ws_size,
                              hipStream_t stream)
{
    const float* x        = (const float*)d_in[0];   // [T,H]
    const float* gate_w   = (const float*)d_in[1];   // [E,H]
    const float* e_bias   = (const float*)d_in[2];   // [E]
    const float* w_gu     = (const float*)d_in[3];   // [E,2I,H]
    const float* w_dn     = (const float*)d_in[4];   // [E,H,I]
    const float* sw_gu    = (const float*)d_in[5];   // [2I,H]
    const float* sw_dn    = (const float*)d_in[6];   // [H,I]
    float* out            = (float*)d_out;           // [T,H]

    // ---- workspace carve-up (all 256B aligned) ----
    uint8_t* w = (uint8_t*)d_ws;
    auto take = [&](size_t bytes) { uint8_t* p = w; w += (bytes + 255) & ~(size_t)255; return p; };
    int*    topk_ids = (int*)    take((size_t)kT * kTopK * sizeof(int));
    float*  topk_w   = (float*)  take((size_t)kT * kTopK * sizeof(float));
    int*    counts   = (int*)    take((size_t)kE * sizeof(int));
    int*    list_t   = (int*)    take((size_t)kE * kT * sizeof(int));
    int*    list_k   = (int*)    take((size_t)kE * kT * sizeof(int));
    __bf16* act_r    = (__bf16*) take((size_t)kT * kTopK * kI * sizeof(__bf16));
    __bf16* act_s    = (__bf16*) take((size_t)kT * kI * sizeof(__bf16));
    float*  partial  = (float*)  take((size_t)kT * kTopK * kH * sizeof(float));
    float*  sh_out   = (float*)  take((size_t)kT * kH * sizeof(float));

    // 1) router (also zeroes counts)
    router_kernel<<<kT / 4, 128, 0, stream>>>(x, gate_w, e_bias, topk_ids, topk_w, counts);

    // 2) compact per-expert lists
    compact_kernel<<<kT / 256, 256, 0, stream>>>(topk_ids, counts, list_t, list_k);

    // 3) gate_up + SiLU*mul : routed then shared
    gu_kernel<<<dim3(kI / 128, kT / 32, kE), 256, 0, stream>>>(
        x, w_gu, list_t, list_k, counts, act_r, 0);
    gu_kernel<<<dim3(kI / 128, kT / 32, 1), 256, 0, stream>>>(
        x, sw_gu, list_t, list_k, counts, act_s, 1);

    // 4) down projection : routed then shared
    down_kernel<<<dim3(kH / 128, kT / 32, kE), 256, 0, stream>>>(
        act_r, w_dn, list_t, list_k, counts, partial, 0);
    down_kernel<<<dim3(kH / 128, kT / 32, 1), 256, 0, stream>>>(
        act_s, sw_dn, list_t, list_k, counts, sh_out, 1);

    // 5) deterministic combine
    combine_kernel<<<(kT * kH) / 256, 256, 0, stream>>>(sh_out, partial, topk_w, out);

    (void)in_sizes; (void)n_in; (void)out_size; (void)ws_size;
}
